// MultiHeadAttention_Encoder_13615046328413
// MI455X (gfx1250) — compile-verified, matrix operand order checked by reading, not run on hardware
//
#include <hip/hip_runtime.h>
#include <hip/hip_bf16.h>

// ---------------------------------------------------------------------------
// MultiHeadAttention encoder for MI455X (gfx1250), all matmuls on
// v_wmma_f32_16x16x32_bf16, flash-style fused softmax (no SxS tensor).
// All WMMA fragments load as 2x b128; GEMMs are double-buffered (register
// prefetch of the next k-step tile overlaps global latency with WMMA).
// B=4, S=1024, D=1024, H=16, hd=64.
// ---------------------------------------------------------------------------

typedef __attribute__((ext_vector_type(16))) __bf16 v16bf;
typedef __attribute__((ext_vector_type(8)))  float  v8f;

#define NB   4
#define NS   1024
#define ND   1024
#define NH   16
#define NHD  64
#define NQKV 3072
#define NEG_BIG (-3.4028235e38f)

union Frag16 {
  v16bf v;
  uint4 q[2];
};

__device__ __forceinline__ v8f wmma_bf16(v16bf a, v16bf b, v8f c) {
  return __builtin_amdgcn_wmma_f32_16x16x32_bf16(
      /*neg_a=*/false, a, /*neg_b=*/false, b,
      /*c_mod=*/(short)0, c, /*reuse_a=*/false, /*reuse_b=*/false);
}

// Per ISA 7.12.2, a lane's 16 elements of a 16-bit 16x32 fragment are two
// contiguous 8-element runs: k = half*8 + [0..7] and k = 16 + half*8 + [0..7].
// `rowbase` points at the 32-element K-run of this lane's fragment row.
// Requires 16B alignment of rowbase + half*8 (stride % 8 elems == 0).
__device__ __forceinline__ v16bf load_frag(const __bf16* rowbase, int half) {
  Frag16 f;
  f.q[0] = *(const uint4*)(rowbase + half * 8);
  f.q[1] = *(const uint4*)(rowbase + 16 + half * 8);
  return f.v;
}

// ---------------------------------------------------------------------------
// 1) f32 -> bf16 conversion
// ---------------------------------------------------------------------------
__global__ __launch_bounds__(256)
void cvt_f32_bf16(const float* __restrict__ src, __bf16* __restrict__ dst, int n) {
  int i = (blockIdx.x * 256 + threadIdx.x) * 4;
  if (i + 3 < n) {
    float4 v = *(const float4*)(src + i);
    dst[i + 0] = (__bf16)v.x;
    dst[i + 1] = (__bf16)v.y;
    dst[i + 2] = (__bf16)v.z;
    dst[i + 3] = (__bf16)v.w;
  }
}

// ---------------------------------------------------------------------------
// Shared GEMM mainloop: 128x64 block tile, K in steps of 32, 256 thr = 8 waves.
// A: MxK row-major bf16. Bw: KxN row-major bf16, staged TRANSPOSED into LDS
// (Bs[n][k], stride 40) so B fragments are two b128 loads per lane.
// Wave w owns output rows w*16..w*16+15; B staging done by waves 0..3
// (wave-uniform branch, EXEC stays all-ones per wave as WMMA requires).
// Double-buffered: next k-step fetched into registers during compute.
// ---------------------------------------------------------------------------
__device__ __forceinline__ void gemm_mainloop_128x64(
    const __bf16* __restrict__ A, const __bf16* __restrict__ Bw,
    int K, int N, int m0, int n0,
    __bf16* As /*128 x stride40*/, __bf16* Bs /*64 x stride40, [n][k]*/,
    v8f acc[4]) {
  int tid  = threadIdx.x;
  int wave = tid >> 5, lane = tid & 31;
  int col16 = lane & 15, half = lane >> 4;
  int rA0 = tid >> 2, gA = tid & 3;        // A slots: rows 0..63 / 64..127
  int rA1 = rA0 + 64;
  int kp = tid >> 3, gB = tid & 7;         // B staging (waves 0..3 only)
  bool doB = wave < 4;

  uint4 a0, a1, b0 = {}, b1 = {};
  a0 = *(const uint4*)(&A[(size_t)(m0 + rA0) * K + gA * 8]);
  a1 = *(const uint4*)(&A[(size_t)(m0 + rA1) * K + gA * 8]);
  if (doB) {
    b0 = *(const uint4*)(&Bw[(size_t)(2 * kp) * N + n0 + gB * 8]);
    b1 = *(const uint4*)(&Bw[(size_t)(2 * kp + 1) * N + n0 + gB * 8]);
  }

  for (int k0 = 0; k0 < K; k0 += 32) {
    // ---- commit staged tiles to LDS
    *(uint4*)(&As[rA0 * 40 + gA * 8]) = a0;
    *(uint4*)(&As[rA1 * 40 + gA * 8]) = a1;
    if (doB) {
      union { uint4 q; unsigned short u[8]; } x0, x1;
      x0.q = b0; x1.q = b1;
#pragma unroll
      for (int e = 0; e < 8; ++e) {
        unsigned int pk = (unsigned int)x0.u[e] | ((unsigned int)x1.u[e] << 16);
        *(unsigned int*)(&Bs[(gB * 8 + e) * 40 + 2 * kp]) = pk;
      }
    }
    __syncthreads();
    // ---- prefetch next k-step (clamped: last iteration refetches harmlessly)
    int kn = (k0 + 32 < K) ? (k0 + 32) : k0;
    a0 = *(const uint4*)(&A[(size_t)(m0 + rA0) * K + kn + gA * 8]);
    a1 = *(const uint4*)(&A[(size_t)(m0 + rA1) * K + kn + gA * 8]);
    if (doB) {
      b0 = *(const uint4*)(&Bw[(size_t)(kn + 2 * kp) * N + n0 + gB * 8]);
      b1 = *(const uint4*)(&Bw[(size_t)(kn + 2 * kp + 1) * N + n0 + gB * 8]);
    }
    // ---- compute
    v16bf a = load_frag(&As[(wave * 16 + col16) * 40], half);
#pragma unroll
    for (int f = 0; f < 4; ++f) {
      v16bf b = load_frag(&Bs[(f * 16 + col16) * 40], half);
      acc[f] = wmma_bf16(a, b, acc[f]);
    }
    __syncthreads();
  }
}

// ---------------------------------------------------------------------------
// 2) QKV GEMM: x(4096x1024) @ Wqkv(1024x3072), epilogue scatters into
//    Q/K [b][h][s][64] and V TRANSPOSED [b][h][64][s] (reproducing the
//    reference's reshape scramble). Q pre-scaled by 1/sqrt(64).
// ---------------------------------------------------------------------------
__global__ __launch_bounds__(256)
void qkv_gemm(const __bf16* __restrict__ A, const __bf16* __restrict__ Bw,
              __bf16* __restrict__ Qb, __bf16* __restrict__ Kb,
              __bf16* __restrict__ Vt) {
  __shared__ __align__(16) __bf16 As[128 * 40];
  __shared__ __align__(16) __bf16 Bs[64 * 40];
  int wave = threadIdx.x >> 5, lane = threadIdx.x & 31;
  int m0 = blockIdx.y * 128, n0 = blockIdx.x * 64;
  v8f acc[4] = {};
  gemm_mainloop_128x64(A, Bw, ND, NQKV, m0, n0, As, Bs, acc);

  int col16 = lane & 15, half = lane >> 4;
#pragma unroll
  for (int f = 0; f < 4; ++f) {
    int n = n0 + f * 16 + col16;
    int j = n / 192, c = n % 192;       // head slot / channel within (q,k,v)
    int kind = c >> 6, d0 = c & 63;
#pragma unroll
    for (int r = 0; r < 8; ++r) {
      int row = m0 + wave * 16 + r + half * 8;   // = b*1024 + s_o
      int b  = row >> 10, s_o = row & 1023;
      int t  = s_o * 16 + j;                     // reshape scramble
      int h  = t >> 10, s = t & 1023;
      size_t bh = (size_t)b * NH + h;
      float v = acc[f][r];
      if (kind == 0)      Qb[(bh * NS + s) * NHD + d0] = (__bf16)(v * 0.125f);
      else if (kind == 1) Kb[(bh * NS + s) * NHD + d0] = (__bf16)v;
      else                Vt[(bh * NHD + d0) * NS + s] = (__bf16)v;
    }
  }
}

// ---------------------------------------------------------------------------
// 3) Flash attention: one wave per (b, h, 16-row q-tile). Streaming softmax,
//    column/row padding masks, writes bf16 `values` in the reshaped layout.
//    K is [kv][hd] (== B-frag [n][k] layout); V is pre-transposed [hd][kv].
// ---------------------------------------------------------------------------
__global__ __launch_bounds__(32)
void attn_kernel(const __bf16* __restrict__ Qb, const __bf16* __restrict__ Kb,
                 const __bf16* __restrict__ Vt,
                 const unsigned char* __restrict__ mask,
                 __bf16* __restrict__ valuesb) {
  __shared__ __align__(16) __bf16 Pl[16 * 40];   // P tile, C-layout -> A-layout
  int lane = threadIdx.x;
  int qt = blockIdx.x & 63, bh = blockIdx.x >> 6;
  int b = bh >> 4, h = bh & 15;
  const __bf16* Qp = Qb + ((size_t)bh * NS) * NHD;
  const __bf16* Kp = Kb + ((size_t)bh * NS) * NHD;
  const __bf16* Vp = Vt + ((size_t)bh * NHD) * NS;
  const unsigned char* mrow = mask + b * NS;
  int q0 = qt * 16;
  int col16 = lane & 15, half = lane >> 4;

  v16bf aQ[2];
#pragma unroll
  for (int t = 0; t < 2; ++t)
    aQ[t] = load_frag(Qp + (size_t)(q0 + col16) * NHD + t * 32, half);

  float m8[8], l8[8];
  v8f o[4] = {};
#pragma unroll
  for (int r = 0; r < 8; ++r) { m8[r] = -INFINITY; l8[r] = 0.f; }

  for (int kb = 0; kb < NS; kb += 32) {
    // ---- scores: two 16x16 tiles, contraction over hd=64 (2 WMMAs each)
    v8f s[2] = {};
#pragma unroll
    for (int t = 0; t < 2; ++t) {
#pragma unroll
      for (int u = 0; u < 2; ++u) {
        v16bf bK = load_frag(Kp + (size_t)(kb + t * 16 + col16) * NHD + u * 32, half);
        s[t] = wmma_bf16(aQ[u], bK, s[t]);
      }
      if (!mrow[kb + t * 16 + col16]) {        // column padding mask
#pragma unroll
        for (int r = 0; r < 8; ++r) s[t][r] = NEG_BIG;
      }
    }
    // ---- online softmax (each row's 16 cols live in one VGPR across a half)
    float corr[8];
#pragma unroll
    for (int r = 0; r < 8; ++r) {
      float sm = fmaxf(s[0][r], s[1][r]);
      sm = fmaxf(sm, __shfl_xor(sm, 1, 16));
      sm = fmaxf(sm, __shfl_xor(sm, 2, 16));
      sm = fmaxf(sm, __shfl_xor(sm, 4, 16));
      sm = fmaxf(sm, __shfl_xor(sm, 8, 16));
      float mn = fmaxf(m8[r], sm);
      corr[r] = __expf(m8[r] - mn);
      m8[r] = mn;
      s[0][r] = __expf(s[0][r] - mn);
      s[1][r] = __expf(s[1][r] - mn);
      float rs = s[0][r] + s[1][r];
      rs += __shfl_xor(rs, 1, 16);
      rs += __shfl_xor(rs, 2, 16);
      rs += __shfl_xor(rs, 4, 16);
      rs += __shfl_xor(rs, 8, 16);
      l8[r] = l8[r] * corr[r] + rs;
    }
#pragma unroll
    for (int f = 0; f < 4; ++f)
#pragma unroll
      for (int r = 0; r < 8; ++r) o[f][r] *= corr[r];

    // ---- P (16x32) through LDS: C-layout -> A-layout
#pragma unroll
    for (int t = 0; t < 2; ++t)
#pragma unroll
      for (int r = 0; r < 8; ++r)
        Pl[(r + half * 8) * 40 + t * 16 + col16] = (__bf16)s[t][r];
    asm volatile("s_wait_dscnt 0" ::: "memory");
    v16bf aP = load_frag(&Pl[col16 * 40], half);

    // ---- O += P @ V  (4 hd chunks of 16), V transposed: [hd][kv]
#pragma unroll
    for (int f = 0; f < 4; ++f) {
      v16bf bV = load_frag(Vp + (size_t)(f * 16 + col16) * NS + kb, half);
      o[f] = wmma_bf16(aP, bV, o[f]);
    }
  }

  // ---- normalize, row mask, scatter into reshaped values layout
#pragma unroll
  for (int r = 0; r < 8; ++r) {
    int sidx = q0 + r + half * 8;
    float scale = mrow[sidx] ? (1.f / l8[r]) : 0.f;   // row padding mask
    int sp = h * 64 + (sidx >> 4);                    // values reshape scramble
    int cp = (sidx & 15) * 64;
#pragma unroll
    for (int f = 0; f < 4; ++f) {
      int hd = f * 16 + col16;
      valuesb[((size_t)b * NS + sp) * ND + cp + hd] = (__bf16)(o[f][r] * scale);
    }
  }
}

// ---------------------------------------------------------------------------
// 4) Output projection: values(4096x1024) @ Wo(1024x1024) + bo -> f32 out
// ---------------------------------------------------------------------------
__global__ __launch_bounds__(256)
void proj_gemm(const __bf16* __restrict__ A, const __bf16* __restrict__ Bw,
               const float* __restrict__ bias, float* __restrict__ out) {
  __shared__ __align__(16) __bf16 As[128 * 40];
  __shared__ __align__(16) __bf16 Bs[64 * 40];
  int wave = threadIdx.x >> 5, lane = threadIdx.x & 31;
  int m0 = blockIdx.y * 128, n0 = blockIdx.x * 64;
  v8f acc[4] = {};
  gemm_mainloop_128x64(A, Bw, ND, ND, m0, n0, As, Bs, acc);

  int col16 = lane & 15, half = lane >> 4;
#pragma unroll
  for (int f = 0; f < 4; ++f) {
    int n = n0 + f * 16 + col16;
    float bn = bias[n];
#pragma unroll
    for (int r = 0; r < 8; ++r) {
      int row = m0 + wave * 16 + r + half * 8;
      out[(size_t)row * ND + n] = acc[f][r] + bn;
    }
  }
}

// ---------------------------------------------------------------------------
extern "C" void kernel_launch(void* const* d_in, const int* in_sizes, int n_in,
                              void* d_out, int out_size, void* d_ws, size_t ws_size,
                              hipStream_t stream) {
  (void)in_sizes; (void)n_in; (void)out_size; (void)ws_size;
  const float*         x    = (const float*)d_in[0];
  const unsigned char* mask = (const unsigned char*)d_in[1];
  const float*         Wqkv = (const float*)d_in[2];
  const float*         Wo   = (const float*)d_in[3];
  const float*         bo   = (const float*)d_in[4];
  float* out = (float*)d_out;

  char* ws = (char*)d_ws;
  const size_t MiB = 1024ull * 1024ull;
  __bf16* xb      = (__bf16*)(ws);              //  8 MiB : 4096x1024
  __bf16* wqkvb   = (__bf16*)(ws +  8 * MiB);   //  6 MiB : 1024x3072
  __bf16* wob     = (__bf16*)(ws + 14 * MiB);   //  2 MiB : 1024x1024
  __bf16* Qb      = (__bf16*)(ws + 16 * MiB);   //  8 MiB : [b][h][s][64]
  __bf16* Kb      = (__bf16*)(ws + 24 * MiB);   //  8 MiB : [b][h][s][64]
  __bf16* Vt      = (__bf16*)(ws + 32 * MiB);   //  8 MiB : [b][h][64][s]
  __bf16* valuesb = (__bf16*)(ws + 40 * MiB);   //  8 MiB : 4096x1024

  cvt_f32_bf16<<<4096, 256, 0, stream>>>(x,    xb,    NB * NS * ND);
  cvt_f32_bf16<<<3072, 256, 0, stream>>>(Wqkv, wqkvb, ND * NQKV);
  cvt_f32_bf16<<<1024, 256, 0, stream>>>(Wo,   wob,   ND * ND);

  qkv_gemm<<<dim3(NQKV / 64, (NB * NS) / 128), 256, 0, stream>>>(xb, wqkvb, Qb, Kb, Vt);
  attn_kernel<<<NB * NH * (NS / 16), 32, 0, stream>>>(Qb, Kb, Vt, mask, valuesb);
  proj_gemm<<<dim3(ND / 64, (NB * NS) / 128), 256, 0, stream>>>(valuesb, wob, bo, out);
}
